// SingleHeadAttention_70325794505147
// MI455X (gfx1250) — matrix operand order repaired and checked by reading, not compiled, not run on hardware
//
#include <hip/hip_runtime.h>

// CDNA5 / gfx1250 WMMA types
typedef __attribute__((ext_vector_type(16))) __bf16 v16bf;
typedef __attribute__((ext_vector_type(8)))  float  v8f;
typedef __attribute__((ext_vector_type(4)))  float  v4f;
typedef __attribute__((ext_vector_type(2)))  unsigned int v2u;
typedef __attribute__((ext_vector_type(4)))  int    v4i;

#if defined(__AMDGCN__) && __has_builtin(__builtin_amdgcn_global_load_async_to_lds_b128)
#define HAVE_ASYNC 1
#else
#define HAVE_ASYNC 0
#endif

__device__ __forceinline__ unsigned short bf16u(float f) {
  union { __bf16 b; unsigned short u; } p; p.b = (__bf16)f; return p.u;
}
__device__ __forceinline__ unsigned int pk_bf16(float lo, float hi) {
  union { __bf16 b[2]; unsigned int u; } p;
  p.b[0] = (__bf16)lo; p.b[1] = (__bf16)hi; return p.u;   // -> v_cvt_pk_bf16_f32
}

__device__ __forceinline__ void wait_async_lds() {
#if HAVE_ASYNC
#if __has_builtin(__builtin_amdgcn_s_wait_asynccnt)
  __builtin_amdgcn_s_wait_asynccnt(0);
#else
  asm volatile("s_wait_asynccnt 0x0" ::: "memory");
#endif
#endif
}

union FragBF16 { unsigned int u[8]; v16bf v; };

#define BM 128
#define BN 128
#define BK 32
#define LDSK 40   // padded element stride (even -> aligned bf16 pairs; 80B rows)

// Stage one BMxBK (rowsxK, NT) or BKxBN (NN source) tile into Tile[r*LDSK+k] (bf16).
template<bool F32, bool NT>
__device__ __forceinline__ void stage_tile(const void* __restrict__ Sp,
                                           unsigned short* Tile,
                                           int base, int ld, int k0, int tid)
{
  if (NT) {
#if HAVE_ASYNC
    if (!F32) {
      // direct global->LDS async DMA, 16B per lane per op (ASYNCcnt-tracked)
      #pragma unroll
      for (int i = 0; i < 2; ++i) {
        int id = i * 256 + tid;        // 512 x 16B chunks = 128 rows x 64B
        int r  = id >> 2;
        int cb = (id & 3) * 16;
        const char* g = (const char*)Sp + ((size_t)(base + r) * ld + k0) * 2 + cb;
        char* l = (char*)&Tile[r * LDSK] + cb;
        __builtin_amdgcn_global_load_async_to_lds_b128(
            (__attribute__((address_space(1))) v4i*)g,
            (__attribute__((address_space(3))) v4i*)l, 0, 0);
      }
      return;
    }
#endif
    #pragma unroll
    for (int i = 0; i < 4; ++i) {
      int idx = (i * 256 + tid) * 4;
      int r = idx >> 5, c = idx & 31;          // 4 consecutive K elems / thread
      v2u d;
      if (F32) {
        v4f g = *(const v4f*)((const float*)Sp + (size_t)(base + r) * ld + k0 + c);
        d.x = pk_bf16(g.x, g.y);
        d.y = pk_bf16(g.z, g.w);
      } else {
        d = *(const v2u*)((const unsigned short*)Sp + (size_t)(base + r) * ld + k0 + c);
      }
      *(v2u*)&Tile[r * LDSK + c] = d;          // ds_store_b64
    }
  } else {
    // source stored [K,N]; Tile[n][k] -- 4 consecutive n per thread, scatter in k col
    #pragma unroll
    for (int i = 0; i < 4; ++i) {
      int idx = (i * 256 + tid) * 4;
      int n = idx & 127, k = idx >> 7;
      unsigned short h[4];
      if (F32) {
        v4f g = *(const v4f*)((const float*)Sp + (size_t)(k0 + k) * ld + base + n);
        h[0] = bf16u(g.x); h[1] = bf16u(g.y); h[2] = bf16u(g.z); h[3] = bf16u(g.w);
      } else {
        v2u d = *(const v2u*)((const unsigned short*)Sp + (size_t)(k0 + k) * ld + base + n);
        h[0] = (unsigned short)(d.x & 0xFFFFu); h[1] = (unsigned short)(d.x >> 16);
        h[2] = (unsigned short)(d.y & 0xFFFFu); h[3] = (unsigned short)(d.y >> 16);
      }
      #pragma unroll
      for (int j = 0; j < 4; ++j) Tile[(n + j) * LDSK + k] = h[j];
    }
  }
}

// C[M,N] = scale * A[M,K] x B;  B_NT: B stored [N,K] (y=A@B^T), else B stored [K,N].
template<bool A_F32, bool B_F32, bool B_NT, bool OUT_BF16, bool CAUSAL>
__global__ __launch_bounds__(256)
void gemm_bf16_wmma(const void* __restrict__ Ap, const void* __restrict__ Bp,
                    void* __restrict__ Cp, int M, int N, int K,
                    int lda, int ldb, int ldc, float scale)
{
  __shared__ unsigned short As[2][BM * LDSK];
  __shared__ unsigned short Bs[2][BN * LDSK];

  const int tid   = threadIdx.x;
  const int mBase = blockIdx.y * BM;
  const int nBase = blockIdx.x * BN;

  if (CAUSAL && nBase > mBase + (BM - 1)) {   // fully-masked block
    float* C = (float*)Cp;
    for (int i = tid; i < BM * BN; i += 256) {
      int r = i >> 7, c = i & (BN - 1);
      C[(size_t)(mBase + r) * ldc + (nBase + c)] = -__builtin_inff();
    }
    return;
  }

  const int lane  = tid & 31;
  const int wave  = tid >> 5;
  const int wm    = wave >> 1;
  const int wn    = wave & 1;
  const int mWave = wm * 32;
  const int nWave = wn * 64;

  const bool useAsync = HAVE_ASYNC && (!A_F32 || (B_NT && !B_F32));

  const v8f zero = {};
  v8f acc[2][4];
  #pragma unroll
  for (int i = 0; i < 2; ++i)
    #pragma unroll
    for (int f = 0; f < 4; ++f) acc[i][f] = zero;

  // per-lane fragment coordinates (ISA 7.12.2, 16-bit layouts)
  const int mA  = lane & 15;
  const int kbA = (lane & 16) ? 8 : 0;
  const int nB  = lane & 15;
  const int kbB = (lane & 16) ? 16 : 0;

  // ---- prologue: stage tile 0 into buffer 0 ----
  stage_tile<A_F32, true>(Ap, As[0], mBase, lda, 0, tid);
  stage_tile<B_F32, B_NT>(Bp, Bs[0], nBase, ldb, 0, tid);
  if (useAsync) wait_async_lds();
  __syncthreads();

  int buf = 0;
  for (int k0 = 0; k0 < K; k0 += BK) {
    const bool hasNext = (k0 + BK) < K;
    if (hasNext) {   // stage next tile into the other buffer while computing
      stage_tile<A_F32, true>(Ap, As[buf ^ 1], mBase, lda, k0 + BK, tid);
      stage_tile<B_F32, B_NT>(Bp, Bs[buf ^ 1], nBase, ldb, k0 + BK, tid);
    }

    const unsigned short* Asb = As[buf];
    const unsigned short* Bsb = Bs[buf];
    FragBF16 a[2], b[4];
    #pragma unroll
    for (int v = 0; v < 8; ++v) {
      int kA = (v < 4) ? (kbA + 2 * v) : (16 + kbA + 2 * (v - 4));
      a[0].u[v] = *(const unsigned int*)&Asb[(mWave +      mA) * LDSK + kA];
      a[1].u[v] = *(const unsigned int*)&Asb[(mWave + 16 + mA) * LDSK + kA];
      int kB = kA;
      #pragma unroll
      for (int f = 0; f < 4; ++f)
        b[f].u[v] = *(const unsigned int*)&Bsb[(nWave + f * 16 + nB) * LDSK + kB];
    }

    #pragma unroll
    for (int i = 0; i < 2; ++i)
      #pragma unroll
      for (int f = 0; f < 4; ++f)
        acc[i][f] = __builtin_amdgcn_wmma_f32_16x16x32_bf16(
            false, a[i].v, false, b[f].v, (short)0, acc[i][f], false, false);

    if (hasNext && useAsync) wait_async_lds();
    __syncthreads();
    buf ^= 1;
  }

  // ---- epilogue: C/D lane layout -> global ----
  #pragma unroll
  for (int i = 0; i < 2; ++i) {
    #pragma unroll
    for (int f = 0; f < 4; ++f) {
      #pragma unroll
      for (int v = 0; v < 8; ++v) {
        int rg = mBase + mWave + i * 16 + v + ((lane & 16) ? 8 : 0);
        int cg = nBase + nWave + f * 16 + (lane & 15);
        float val = acc[i][f][v] * scale;
        if (CAUSAL && cg > rg) val = -__builtin_inff();
        if (OUT_BF16) ((unsigned short*)Cp)[(size_t)rg * ldc + cg] = bf16u(val);
        else          ((float*)Cp)[(size_t)rg * ldc + cg] = val;
      }
    }
  }
}

// Row softmax over fp32 scores -> bf16 probabilities (vectorized)
__global__ __launch_bounds__(256)
void softmax_rows(const float* __restrict__ S, unsigned short* __restrict__ P, int N)
{
  __shared__ float red[8];
  const int row  = blockIdx.x;
  const float* r = S + (size_t)row * N;
  const int tid = threadIdx.x, lane = tid & 31, wave = tid >> 5;

  float m = -__builtin_inff();
  for (int c = tid * 4; c < N; c += 1024) {
    v4f g = *(const v4f*)&r[c];
    m = fmaxf(m, fmaxf(fmaxf(g.x, g.y), fmaxf(g.z, g.w)));
  }
  #pragma unroll
  for (int o = 16; o >= 1; o >>= 1) m = fmaxf(m, __shfl_xor(m, o, 32));
  if (lane == 0) red[wave] = m;
  __syncthreads();
  m = red[lane & 7];
  #pragma unroll
  for (int o = 4; o >= 1; o >>= 1) m = fmaxf(m, __shfl_xor(m, o, 32));
  __syncthreads();

  float s = 0.f;
  for (int c = tid * 4; c < N; c += 1024) {
    v4f g = *(const v4f*)&r[c];
    s += __expf(g.x - m) + __expf(g.y - m) + __expf(g.z - m) + __expf(g.w - m);
  }
  #pragma unroll
  for (int o = 16; o >= 1; o >>= 1) s += __shfl_xor(s, o, 32);
  if (lane == 0) red[wave] = s;
  __syncthreads();
  s = red[lane & 7];
  #pragma unroll
  for (int o = 4; o >= 1; o >>= 1) s += __shfl_xor(s, o, 32);

  float inv = 1.0f / s;
  for (int c = tid * 4; c < N; c += 1024) {
    v4f g = *(const v4f*)&r[c];
    v2u o2;
    o2.x = pk_bf16(__expf(g.x - m) * inv, __expf(g.y - m) * inv);
    o2.y = pk_bf16(__expf(g.z - m) * inv, __expf(g.w - m) * inv);
    *(v2u*)&P[(size_t)row * N + c] = o2;
  }
}

extern "C" void kernel_launch(void* const* d_in, const int* in_sizes, int n_in,
                              void* d_out, int out_size, void* d_ws, size_t ws_size,
                              hipStream_t stream)
{
  const int Bb = 4, S = 4096, D = 1024;
  const float qscale = 0.03125f;  // 1/sqrt(1024)

  const float* x  = (const float*)d_in[0];
  const float* Wq = (const float*)d_in[1];
  const float* Wk = (const float*)d_in[2];
  const float* Wv = (const float*)d_in[3];
  const float* Wo = (const float*)d_in[4];
  float* out = (float*)d_out;

  // ws: Q|K|V|context bf16 (B*S*D each) + fp32 scores S*S + bf16 probs S*S
  size_t SD = (size_t)Bb * S * D;
  unsigned short* Qb = (unsigned short*)d_ws;
  unsigned short* Kb = Qb + SD;
  unsigned short* Vb = Kb + SD;
  unsigned short* Cb = Vb + SD;
  float* Sc          = (float*)(Cb + SD);
  unsigned short* Pb = (unsigned short*)(Sc + (size_t)S * S);

  dim3 blk(256);

  dim3 gQKV(D / BN, (Bb * S) / BM);
  gemm_bf16_wmma<true, true, true, true, false><<<gQKV, blk, 0, stream>>>(
      x, Wq, Qb, Bb * S, D, D, D, D, D, qscale);
  gemm_bf16_wmma<true, true, true, true, false><<<gQKV, blk, 0, stream>>>(
      x, Wk, Kb, Bb * S, D, D, D, D, D, 1.0f);
  gemm_bf16_wmma<true, true, true, true, false><<<gQKV, blk, 0, stream>>>(
      x, Wv, Vb, Bb * S, D, D, D, D, D, 1.0f);

  for (int b = 0; b < Bb; ++b) {
    const unsigned short* Qp = Qb + (size_t)b * S * D;
    const unsigned short* Kp = Kb + (size_t)b * S * D;
    const unsigned short* Vp = Vb + (size_t)b * S * D;
    unsigned short*       Cp = Cb + (size_t)b * S * D;

    dim3 gS(S / BN, S / BM);
    gemm_bf16_wmma<false, false, true, false, true><<<gS, blk, 0, stream>>>(
        Qp, Kp, Sc, S, S, D, D, D, S, 1.0f);

    softmax_rows<<<S, blk, 0, stream>>>(Sc, Pb, S);

    dim3 gC(D / BN, S / BM);
    gemm_bf16_wmma<false, false, false, true, false><<<gC, blk, 0, stream>>>(
        Pb, Vp, Cp, S, D, S, S, D, D, 1.0f);
  }

  dim3 gO(D / BN, (Bb * S) / BM);
  gemm_bf16_wmma<false, true, true, false, false><<<gO, blk, 0, stream>>>(
      Cb, Wo, out, Bb * S, D, D, D, D, D, 1.0f);
}
